// SFFA_81827716923690
// MI455X (gfx1250) — matrix operand order repaired and checked by reading, not compiled, not run on hardware
//
#include <hip/hip_runtime.h>

// Problem constants (reference shapes).
#define BZ   4
#define NC   128        // channels / head dim
#define HW   4096       // N = 64*64 pixels
#define QT   64         // queries per workgroup (4 waves x 16)
#define KBLK 64         // keys per main-loop iteration (2x WMMA K)
#define MAXP 4          // split-K partitions

typedef __attribute__((ext_vector_type(16))) __bf16 v16bf;
typedef __attribute__((ext_vector_type(8)))  float  v8f;

#if __has_builtin(__builtin_amdgcn_global_load_async_to_lds_b128) && \
    __has_builtin(__builtin_amdgcn_s_wait_asynccnt)
#define USE_ASYNC 1
#define AS1 __attribute__((address_space(1)))
#define AS3 __attribute__((address_space(3)))
typedef int vsi4 __attribute__((vector_size(4 * sizeof(int))));
#else
#define USE_ASYNC 0
#endif

// ---------------------------------------------------------------------------
// 16-lane reductions via DPP (pure VALU, co-executes with WMMA); fallback shfl.
// ---------------------------------------------------------------------------
#if __has_builtin(__builtin_amdgcn_update_dpp)
__device__ __forceinline__ float red16_max(float x) {
  float y;
  y = __int_as_float(__builtin_amdgcn_update_dpp(0, __float_as_int(x), 0xB1, 0xf, 0xf, true));
  x = fmaxf(x, y);
  y = __int_as_float(__builtin_amdgcn_update_dpp(0, __float_as_int(x), 0x4E, 0xf, 0xf, true));
  x = fmaxf(x, y);
  y = __int_as_float(__builtin_amdgcn_update_dpp(0, __float_as_int(x), 0x141, 0xf, 0xf, true));
  x = fmaxf(x, y);
  y = __int_as_float(__builtin_amdgcn_update_dpp(0, __float_as_int(x), 0x140, 0xf, 0xf, true));
  x = fmaxf(x, y);
  return x;
}
__device__ __forceinline__ float red16_sum(float x) {
  float y;
  y = __int_as_float(__builtin_amdgcn_update_dpp(0, __float_as_int(x), 0xB1, 0xf, 0xf, true));
  x += y;
  y = __int_as_float(__builtin_amdgcn_update_dpp(0, __float_as_int(x), 0x4E, 0xf, 0xf, true));
  x += y;
  y = __int_as_float(__builtin_amdgcn_update_dpp(0, __float_as_int(x), 0x141, 0xf, 0xf, true));
  x += y;
  y = __int_as_float(__builtin_amdgcn_update_dpp(0, __float_as_int(x), 0x140, 0xf, 0xf, true));
  x += y;
  return x;
}
#else
__device__ __forceinline__ float red16_max(float x) {
  x = fmaxf(x, __shfl_xor(x, 1, 32)); x = fmaxf(x, __shfl_xor(x, 2, 32));
  x = fmaxf(x, __shfl_xor(x, 4, 32)); x = fmaxf(x, __shfl_xor(x, 8, 32));
  return x;
}
__device__ __forceinline__ float red16_sum(float x) {
  x += __shfl_xor(x, 1, 32); x += __shfl_xor(x, 2, 32);
  x += __shfl_xor(x, 4, 32); x += __shfl_xor(x, 8, 32);
  return x;
}
#endif

// ---------------------------------------------------------------------------
// Prep: background = in*mask ; K = bg + 1e-7 ; Kn = K/||K||.
// Emits bf16: Q [n][c], Kn [n][c], and V=K channel-major [c][n] so the attn
// kernel's LDS staging is pure row-major block copies (no inner transposes).
// ---------------------------------------------------------------------------
__global__ __launch_bounds__(256) void sffa_prep_kernel(
    const float* __restrict__ in, const float* __restrict__ mask,
    __bf16* __restrict__ Qb, __bf16* __restrict__ Knb, __bf16* __restrict__ Vc)
{
  __shared__ float tile[QT][NC + 1];
  __shared__ float rns[QT], mvs[QT];
  const int b   = blockIdx.y;
  const int p0  = blockIdx.x * QT;
  const int tid = threadIdx.x;

  for (int idx = tid; idx < QT * NC; idx += 256) {
    int c = idx >> 6;
    int p = idx & (QT - 1);
    tile[p][c] = in[(((size_t)b * NC + c) * HW) + p0 + p];
  }
  __syncthreads();

  {
    const int p   = tid >> 2;
    const int sub = tid & 3;
    const float mv = mask[(size_t)b * HW + p0 + p];
    float ss = 0.f;
    for (int c = sub; c < NC; c += 4) {
      float k = tile[p][c] * mv + 1e-7f;
      ss += k * k;
    }
    ss += __shfl_xor(ss, 1, 32);
    ss += __shfl_xor(ss, 2, 32);
    if (sub == 0) { rns[p] = rsqrtf(ss); mvs[p] = mv; }
  }
  __syncthreads();

  // Q / Kn: [n][c] rows, coalesced 16B stores
  for (int idx = tid; idx < QT * (NC / 8); idx += 256) {
    int p  = idx >> 4;
    int c8 = idx & 15;
    float rn = rns[p], mv = mvs[p];
    uint4 uq, un;
    __bf16* pq = (__bf16*)&uq; __bf16* pn = (__bf16*)&un;
    #pragma unroll
    for (int t = 0; t < 8; ++t) {
      float x = tile[p][c8 * 8 + t];
      float k = x * mv + 1e-7f;
      pq[t] = (__bf16)x; pn[t] = (__bf16)(k * rn);
    }
    size_t base = ((size_t)b * HW + p0 + p) * NC + c8 * 8;
    *(uint4*)(Qb + base)  = uq;
    *(uint4*)(Knb + base) = un;
  }
  // V: [c][n] channel-major, coalesced 16B stores
  for (int idx = tid; idx < NC * (QT / 8); idx += 256) {
    int c  = idx >> 3;
    int p8 = (idx & 7) * 8;
    uint4 uv; __bf16* pv = (__bf16*)&uv;
    #pragma unroll
    for (int t = 0; t < 8; ++t) {
      int p = p8 + t;
      pv[t] = (__bf16)(tile[p][c] * mvs[p] + 1e-7f);
    }
    *(uint4*)(Vc + ((size_t)b * NC + c) * HW + p0 + p8) = uv;
  }
}

// ---------------------------------------------------------------------------
// Stage one 64-key block: Kn rows [key][c] and V rows [c][key].
// Async (ASYNCcnt DMA) when available, else sync vector copies.
// 16 B128 transfers per thread => ASYNCcnt +16 per wave per block.
// ---------------------------------------------------------------------------
__device__ __forceinline__ void stage_block(
    const __bf16* __restrict__ Kng, const __bf16* __restrict__ Vcg,
    __bf16 (*KsB)[NC + 8], __bf16 (*VtB)[KBLK + 24], int kb, int tid)
{
#if USE_ASYNC
  #pragma unroll
  for (int i = 0; i < 8; ++i) {
    int idx  = tid + i * 128;            // 0..1023
    int row  = idx >> 4;                 // key 0..63
    int col8 = idx & 15;
    AS1 vsi4* g = (AS1 vsi4*)(Kng + (size_t)(kb + row) * NC + col8 * 8);
    AS3 vsi4* l = (AS3 vsi4*)(&KsB[row][col8 * 8]);
    __builtin_amdgcn_global_load_async_to_lds_b128(g, l, 0, 0);
  }
  #pragma unroll
  for (int i = 0; i < 8; ++i) {
    int idx = tid + i * 128;             // 0..1023
    int row = idx >> 3;                  // channel 0..127
    int g8  = (idx & 7) * 8;             // key group
    AS1 vsi4* g = (AS1 vsi4*)(Vcg + (size_t)row * HW + kb + g8);
    AS3 vsi4* l = (AS3 vsi4*)(&VtB[row][g8]);
    __builtin_amdgcn_global_load_async_to_lds_b128(g, l, 0, 0);
  }
#else
  #pragma unroll
  for (int i = 0; i < 8; ++i) {
    int idx  = tid + i * 128;
    int row  = idx >> 4;
    int col8 = idx & 15;
    *(uint4*)(&KsB[row][col8 * 8]) =
        *(const uint4*)(Kng + (size_t)(kb + row) * NC + col8 * 8);
  }
  #pragma unroll
  for (int i = 0; i < 8; ++i) {
    int idx = tid + i * 128;
    int row = idx >> 3;
    int g8  = (idx & 7) * 8;
    *(uint4*)(&VtB[row][g8]) =
        *(const uint4*)(Vcg + (size_t)row * HW + kb + g8);
  }
#endif
}

// ---------------------------------------------------------------------------
// Split-K fused attention, double-buffered LDS staging.
// ---------------------------------------------------------------------------
__global__ __launch_bounds__(128) void sffa_attn_kernel(
    const __bf16* __restrict__ Qb, const __bf16* __restrict__ Knb,
    const __bf16* __restrict__ Vc, float* __restrict__ Op,
    float* __restrict__ Ml)
{
  __shared__ __bf16 Qs[QT][NC + 8];            // 17408 B
  __shared__ __bf16 Ks[2][KBLK][NC + 8];       // 34816 B
  __shared__ __bf16 Vt[2][NC][KBLK + 24];      // 45056 B (stride 88: conflict-free)
  __shared__ __bf16 Ps[4][16][KBLK + 16];      // 10240 B

  const int b    = blockIdx.y;
  const int q0   = blockIdx.x * QT;
  const int part = blockIdx.z;
  const int P    = gridDim.z;
  const int tid  = threadIdx.x;
  const int wave = tid >> 5;
  const int lane = tid & 31;
  const int hi   = lane >> 4;
  const int l16  = lane & 15;

  // ---- stage Q tile [64 q][128 c] ----
  const __bf16* Qg = Qb + ((size_t)b * HW + q0) * NC;
  for (int idx = tid; idx < QT * (NC / 8); idx += 128) {
    int row  = idx >> 4;
    int col8 = idx & 15;
    *(uint4*)(&Qs[row][col8 * 8]) =
        *(const uint4*)(Qg + (size_t)row * NC + col8 * 8);
  }
  __syncthreads();

  // ---- Q fragments (A layout) ----
  v16bf aq[4];
  {
    const int row = wave * 16 + l16;
    #pragma unroll
    for (int d = 0; d < 4; ++d)
      #pragma unroll
      for (int j = 0; j < 16; ++j) {
        int k = d * 32 + (hi ? 8 : 0) + (j & 7) + ((j >> 3) << 4);
        aq[d][j] = Qs[row][k];
      }
  }

  float mrow[8], lrow[8];
  v8f o[8] = {};
  #pragma unroll
  for (int v = 0; v < 8; ++v) { mrow[v] = -3.0e38f; lrow[v] = 0.f; }

  const __bf16* Kng = Knb + (size_t)b * HW * NC;
  const __bf16* Vcg = Vc + (size_t)b * NC * HW;
  const int kspan  = HW / P;
  const int kstart = part * kspan;
  const int kend   = kstart + kspan;

  int cur = 0;
  stage_block(Kng, Vcg, Ks[0], Vt[0], kstart, tid);    // prologue

  for (int kb = kstart; kb < kend; kb += KBLK) {
    const int  nxt       = cur ^ 1;
    const bool have_next = (kb + KBLK < kend);
    if (have_next) {
      stage_block(Kng, Vcg, Ks[nxt], Vt[nxt], kb + KBLK, tid);
#if !USE_ASYNC
      __builtin_prefetch((const char*)(Kng + (size_t)(kb + KBLK) * NC) + tid * 128, 0, 1);
#endif
    }
#if USE_ASYNC
    if (have_next) __builtin_amdgcn_s_wait_asynccnt(16);  // drain older block
    else           __builtin_amdgcn_s_wait_asynccnt(0);
#endif
    __syncthreads();   // current block staged for all waves

    // ---- S = Q . Kn^T : four 16x16 key tiles ----
    v8f s[4] = {};
    #pragma unroll
    for (int d = 0; d < 4; ++d) {
      #pragma unroll
      for (int t = 0; t < 4; ++t) {
        v16bf bb;
        #pragma unroll
        for (int j = 0; j < 16; ++j)
          bb[j] = Ks[cur][t * 16 + l16][d * 32 + (hi ? 8 : 0) + (j & 7) + ((j >> 3) << 4)];
        s[t] = __builtin_amdgcn_wmma_f32_16x16x32_bf16(false, aq[d], false, bb,
                                                       (short)0, s[t], false, false);
      }
    }

    // ---- online softmax over this 64-key block ----
    float mnew[8], scale[8];
    #pragma unroll
    for (int v = 0; v < 8; ++v) {
      float mx = fmaxf(fmaxf(s[0][v], s[1][v]), fmaxf(s[2][v], s[3][v]));
      mx = red16_max(mx);
      mnew[v]  = fmaxf(mrow[v], mx);
      scale[v] = __expf(mrow[v] - mnew[v]);
      mrow[v]  = mnew[v];
    }
    #pragma unroll
    for (int v = 0; v < 8; ++v) {
      float e0 = __expf(s[0][v] - mnew[v]);
      float e1 = __expf(s[1][v] - mnew[v]);
      float e2 = __expf(s[2][v] - mnew[v]);
      float e3 = __expf(s[3][v] - mnew[v]);
      lrow[v] = lrow[v] * scale[v] + red16_sum(e0 + e1 + e2 + e3);
      Ps[wave][v + hi * 8][l16]      = (__bf16)e0;
      Ps[wave][v + hi * 8][16 + l16] = (__bf16)e1;
      Ps[wave][v + hi * 8][32 + l16] = (__bf16)e2;
      Ps[wave][v + hi * 8][48 + l16] = (__bf16)e3;
      #pragma unroll
      for (int jt = 0; jt < 8; ++jt) o[jt][v] *= scale[v];
    }

    // ---- O += P . V ----
    #pragma unroll
    for (int kc = 0; kc < 2; ++kc) {
      v16bf pa;
      #pragma unroll
      for (int j = 0; j < 16; ++j)
        pa[j] = Ps[wave][l16][kc * 32 + (hi ? 8 : 0) + (j & 7) + ((j >> 3) << 4)];
      #pragma unroll
      for (int jt = 0; jt < 8; ++jt) {
        v16bf bv;
        #pragma unroll
        for (int j = 0; j < 16; ++j)
          bv[j] = Vt[cur][jt * 16 + l16][kc * 32 + (hi ? 8 : 0) + (j & 7) + ((j >> 3) << 4)];
        o[jt] = __builtin_amdgcn_wmma_f32_16x16x32_bf16(false, pa, false, bv,
                                                        (short)0, o[jt], false, false);
      }
    }
    __syncthreads();   // done reading buffer `cur`; next iter may overwrite it
    cur = nxt;
  }

  // ---- write partials: O[part][c][q] (32B contiguous per lane), (m,l) ----
  float* Opb = Op + (((size_t)b * P + part) * NC) * HW;
  #pragma unroll
  for (int jt = 0; jt < 8; ++jt) {
    int c = jt * 16 + l16;
    size_t base = (size_t)c * HW + q0 + wave * 16 + hi * 8;
    #pragma unroll
    for (int v = 0; v < 8; ++v) Opb[base + v] = o[jt][v];
  }
  if (l16 == 0) {
    #pragma unroll
    for (int v = 0; v < 8; ++v) {
      int q = q0 + wave * 16 + hi * 8 + v;
      size_t moff = (((size_t)b * P + part) * HW + q) * 2;
      Ml[moff]     = mrow[v];
      Ml[moff + 1] = lrow[v];
    }
  }
}

// ---------------------------------------------------------------------------
// Combine partials + mask blend; fully coalesced float4 in [c][q] layout.
// ---------------------------------------------------------------------------
__global__ __launch_bounds__(256) void sffa_combine_kernel(
    const float* __restrict__ in, const float* __restrict__ mask,
    const float* __restrict__ Op, const float* __restrict__ Ml,
    float* __restrict__ out, int parts)
{
  const int idx = blockIdx.x * 256 + threadIdx.x;   // BZ*NC*HW/4 threads
  const int q4  = (idx & (HW / 4 - 1)) * 4;
  const int c   = (idx >> 10) & (NC - 1);
  const int b   = idx >> 17;

  float gm[4] = {-3.0e38f, -3.0e38f, -3.0e38f, -3.0e38f};
  for (int p = 0; p < parts; ++p) {
    size_t mb = (((size_t)b * parts + p) * HW + q4) * 2;
    float4 a0 = *(const float4*)&Ml[mb];       // m0,l0,m1,l1
    float4 a1 = *(const float4*)&Ml[mb + 4];   // m2,l2,m3,l3
    gm[0] = fmaxf(gm[0], a0.x); gm[1] = fmaxf(gm[1], a0.z);
    gm[2] = fmaxf(gm[2], a1.x); gm[3] = fmaxf(gm[3], a1.z);
  }
  float4 num = {0, 0, 0, 0};
  float  den[4] = {0, 0, 0, 0};
  for (int p = 0; p < parts; ++p) {
    size_t mb = (((size_t)b * parts + p) * HW + q4) * 2;
    float4 a0 = *(const float4*)&Ml[mb];
    float4 a1 = *(const float4*)&Ml[mb + 4];
    float w0 = __expf(a0.x - gm[0]), w1 = __expf(a0.z - gm[1]);
    float w2 = __expf(a1.x - gm[2]), w3 = __expf(a1.z - gm[3]);
    den[0] += a0.y * w0; den[1] += a0.w * w1;
    den[2] += a1.y * w2; den[3] += a1.w * w3;
    float4 o4 = *(const float4*)&Op[(((size_t)b * parts + p) * NC + c) * HW + q4];
    num.x += o4.x * w0; num.y += o4.y * w1; num.z += o4.z * w2; num.w += o4.w * w3;
  }
  float4 mk  = *(const float4*)&mask[(size_t)b * HW + q4];
  size_t off = ((size_t)b * NC + c) * HW + q4;
  float4 xin = *(const float4*)&in[off];
  float4 r;
  r.x = xin.x * mk.x + (num.x / den[0]) * (1.f - mk.x);
  r.y = xin.y * mk.y + (num.y / den[1]) * (1.f - mk.y);
  r.z = xin.z * mk.z + (num.z / den[2]) * (1.f - mk.z);
  r.w = xin.w * mk.w + (num.w / den[3]) * (1.f - mk.w);
  *(float4*)&out[off] = r;
}

// ---------------------------------------------------------------------------
extern "C" void kernel_launch(void* const* d_in, const int* in_sizes, int n_in,
                              void* d_out, int out_size, void* d_ws, size_t ws_size,
                              hipStream_t stream) {
  (void)in_sizes; (void)n_in; (void)out_size;
  const float* in   = (const float*)d_in[0];
  const float* mask = (const float*)d_in[1];
  float*       out  = (float*)d_out;

  const size_t PACK  = (size_t)BZ * HW * NC * sizeof(__bf16);        // 4 MB each
  const size_t OPSZ4 = (size_t)BZ * MAXP * NC * HW * sizeof(float);  // 32 MB
  const size_t MLSZ4 = (size_t)BZ * MAXP * HW * 2 * sizeof(float);
  const size_t NEED4 = 3 * PACK + OPSZ4 + MLSZ4;
  const int parts = (ws_size >= NEED4) ? MAXP : 1;

  char* ws = (char*)d_ws;
  __bf16* Qb  = (__bf16*)(ws);
  __bf16* Knb = (__bf16*)(ws + PACK);
  __bf16* Vc  = (__bf16*)(ws + 2 * PACK);
  float*  Op  = (float*)(ws + 3 * PACK);
  float*  Ml  = (float*)(ws + 3 * PACK + (size_t)BZ * parts * NC * HW * sizeof(float));

  sffa_prep_kernel<<<dim3(HW / QT, BZ), 256, 0, stream>>>(in, mask, Qb, Knb, Vc);
  sffa_attn_kernel<<<dim3(HW / QT, BZ, parts), 128, 0, stream>>>(Qb, Knb, Vc, Op, Ml);
  sffa_combine_kernel<<<(BZ * NC * HW / 4) / 256, 256, 0, stream>>>(in, mask, Op, Ml, out, parts);
}
